// SelfAttention_12661563588803
// MI455X (gfx1250) — matrix operand order repaired and checked by reading, not compiled, not run on hardware
//
#include <hip/hip_runtime.h>

// ---------- types ----------
typedef __bf16 bf16_t;
typedef __attribute__((ext_vector_type(16))) __bf16 v16bf;
typedef __attribute__((ext_vector_type(8)))  float  v8f;
typedef __attribute__((ext_vector_type(4)))  unsigned int u32x4;
typedef __attribute__((ext_vector_type(4)))  int v4i;

union Frag32B { v16bf v; u32x4 u[2]; };
union Pack8   { bf16_t h[8]; u32x4 u; };

#define WMMA_BF16(a, b, c) \
    __builtin_amdgcn_wmma_f32_16x16x32_bf16(false, (a), false, (b), (short)0, (c), false, false)

// ---------- CDNA5 async global->LDS copy (ASYNCcnt-tracked) ----------
#if defined(__has_builtin)
#  if __has_builtin(__builtin_amdgcn_global_load_async_to_lds_b128)
#    define ASYNC_B128_BUILTIN 1
#  endif
#  if __has_builtin(__builtin_amdgcn_s_wait_asynccnt)
#    define WAIT_ASYNC_BUILTIN 1
#  endif
#endif

typedef __attribute__((address_space(1))) v4i gv4i;   // global (AS1, prints as __device__)
typedef __attribute__((address_space(3))) v4i lv4i;   // LDS (AS3)

__device__ __forceinline__ void async_ld_b128(void* lds_dst, const void* gsrc) {
#if defined(ASYNC_B128_BUILTIN)
    gv4i* gp = (gv4i*)(unsigned long long)gsrc;                 // u64 -> AS1 ptr
    lv4i* lp = (lv4i*)(unsigned)(unsigned long long)lds_dst;    // low 32b = LDS offset -> AS3 ptr
    __builtin_amdgcn_global_load_async_to_lds_b128(gp, lp, 0, 0);
#else
    unsigned loff = (unsigned)(unsigned long long)lds_dst;
    unsigned long long ga = (unsigned long long)gsrc;
    asm volatile("global_load_async_to_lds_b128 %0, %1, off"
                 :: "v"(loff), "v"(ga) : "memory");
#endif
}

__device__ __forceinline__ void wait_async0() {
#if defined(WAIT_ASYNC_BUILTIN)
    __builtin_amdgcn_s_wait_asynccnt(0);
#else
    asm volatile("s_wait_asynccnt 0" ::: "memory");
#endif
}

// ---------- fp32 -> bf16 conversion ----------
__global__ __launch_bounds__(256) void cvt_f32_to_bf16(const float* __restrict__ in,
                                                       bf16_t* __restrict__ out, int n) {
    int i = blockIdx.x * 256 + threadIdx.x;
    if (i < n) out[i] = (bf16_t)in[i];
}

// ---------- WMMA GEMM: C[M,N] = A[M,K] @ B[K,N] + bias ----------
// Block: 256 threads (8 waves). Tile: 128(M) x 64(N), K-step 32.
// Wave w computes rows [w*16, w*16+16) x all 64 cols (4 accumulators).
template <bool F32OUT>
__global__ __launch_bounds__(256) void gemm_wmma_bf16(
    const bf16_t* __restrict__ A,   // M x K row-major (bf16)
    const bf16_t* __restrict__ B,   // K x N row-major (bf16)
    const float*  __restrict__ bias,// N
    float*        __restrict__ Cf,  // used when F32OUT
    bf16_t*       __restrict__ Cb,  // used when !F32OUT
    int M, int N, int K)
{
    __shared__ __align__(16) bf16_t As[128][32];  // A tile, row-major
    __shared__ __align__(16) bf16_t Bts[64][32];  // B tile, TRANSPOSED: Bts[n][k]

    const int tid    = threadIdx.x;
    const int wave   = tid >> 5;
    const int lane   = tid & 31;
    const int laneLo = lane & 15;
    const int hi     = (lane >> 4) & 1;

    const int m0 = blockIdx.y * 128;
    const int n0 = blockIdx.x * 64;

    v8f acc[4];
    #pragma unroll
    for (int j = 0; j < 4; ++j) {
        v8f z = {0.f,0.f,0.f,0.f,0.f,0.f,0.f,0.f};
        acc[j] = z;
    }

    // per-thread staging coordinates
    const int ar  = tid >> 1;            // A tile row
    const int ah2 = (tid & 1) * 16;      // A tile k-half
    const int bn  = tid >> 2;            // B tile col
    const int bks = (tid & 3) * 8;       // B tile k-seg

    for (int k0 = 0; k0 < K; k0 += 32) {
        __syncthreads();
        // --- stage A tile with async global->LDS (2 x b128 per thread) ---
        {
            const bf16_t* src = A + (size_t)(m0 + ar) * K + k0 + ah2;
            async_ld_b128(&As[ar][ah2],     src);
            async_ld_b128(&As[ar][ah2 + 8], src + 8);
            if (k0 + 32 < K) __builtin_prefetch(src + 32, 0, 1); // next A tile
        }
        // --- stage B tile transposed: thread -> (col n, 8 consecutive k) ---
        {
            Pack8 pk;
            const bf16_t* bp = B + (size_t)(k0 + bks) * N + (n0 + bn);
            #pragma unroll
            for (int i = 0; i < 8; ++i)
                pk.h[i] = bp[(size_t)i * N];
            *((u32x4*)&Bts[bn][bks]) = pk.u;
            if (k0 + 32 < K) __builtin_prefetch(bp + (size_t)32 * N, 0, 1);
        }
        wait_async0();
        __syncthreads();

        // --- A fragment (16x32): lane = row, K split {0..7,16..23} / {8..15,24..31} ---
        Frag32B fa;
        {
            const bf16_t* p = &As[wave * 16 + laneLo][hi * 8];
            fa.u[0] = *((const u32x4*)p);
            fa.u[1] = *((const u32x4*)(p + 16));
        }
        // --- 4 WMMAs reusing A ---
        #pragma unroll
        for (int j = 0; j < 4; ++j) {
            Frag32B fb;
            const bf16_t* p = &Bts[j * 16 + laneLo][hi * 8];
            fb.u[0] = *((const u32x4*)p);
            fb.u[1] = *((const u32x4*)(p + 16));
            acc[j] = WMMA_BF16(fa.v, fb.v, acc[j]);
        }
    }

    // --- epilogue: C/D layout -> lane&15 = col, VGPR r -> row r + 8*hi ---
    #pragma unroll
    for (int r = 0; r < 8; ++r) {
        const size_t row = (size_t)(m0 + wave * 16 + r + hi * 8);
        #pragma unroll
        for (int j = 0; j < 4; ++j) {
            int col  = n0 + j * 16 + laneLo;
            float v  = acc[j][r] + bias[col];
            if constexpr (F32OUT) Cf[row * N + col] = v;
            else                  Cb[row * N + col] = (bf16_t)v;
        }
    }
}

// ---------- Flash attention ----------
// Grid: (S/128, B*16). Block: 256 threads (8 waves); wave owns 16 query rows.
// K-tile = 32 keys per iteration, online softmax, d = 64.
__global__ __launch_bounds__(256) void attn_flash_wmma(
    const bf16_t* __restrict__ qkv,   // [B*S][3072] = [q(1024) k(1024) v(1024)]
    bf16_t*       __restrict__ attnO, // [B*S][1024]
    int S)
{
    __shared__ __align__(16) bf16_t Ks[32][64];      // K tile row-major -> QK^T B-frags
    __shared__ __align__(16) bf16_t Vts[64][32];     // V tile transposed -> PV B-frags
    __shared__ __align__(16) bf16_t Ps[8][16][32];   // per-wave P scratch (C->A relayout)

    const int tid    = threadIdx.x;
    const int wave   = tid >> 5;
    const int lane   = tid & 31;
    const int laneLo = lane & 15;
    const int hi     = (lane >> 4) & 1;

    const int bh = blockIdx.y;
    const int b  = bh >> 4;
    const int h  = bh & 15;
    const int q0 = blockIdx.x * 128;

    const size_t rs = 3072;
    const bf16_t* Qb = qkv + (size_t)b * S * rs + (size_t)h * 64;
    const bf16_t* Kb = Qb + 1024;
    const bf16_t* Vb = Qb + 2048;

    // Q fragments (reused across all key tiles): rows q0 + wave*16 + laneLo
    Frag32B qf[2];
    {
        const bf16_t* qr = Qb + (size_t)(q0 + wave * 16 + laneLo) * rs;
        #pragma unroll
        for (int kk = 0; kk < 2; ++kk) {
            int d0 = kk * 32 + hi * 8;
            qf[kk].u[0] = *((const u32x4*)(qr + d0));
            qf[kk].u[1] = *((const u32x4*)(qr + d0 + 16));
        }
    }

    v8f o[4];
    #pragma unroll
    for (int j = 0; j < 4; ++j) {
        v8f z = {0.f,0.f,0.f,0.f,0.f,0.f,0.f,0.f};
        o[j] = z;
    }
    float mrow[8], lrow[8];
    #pragma unroll
    for (int r = 0; r < 8; ++r) { mrow[r] = -1e30f; lrow[r] = 0.f; }

    const float scale = 0.125f;          // 1/sqrt(64)
    const float log2e = 1.4426950408889634f;

    // staging coordinates
    const int kkey = tid >> 3;           // key row for K tile
    const int kseg = (tid & 7) * 8;      // d-seg for K tile
    const int vd   = tid >> 2;           // d row for Vt tile
    const int vks  = (tid & 3) * 8;      // key-seg for Vt tile

    for (int kb = 0; kb < S; kb += 32) {
        __syncthreads();
        // stage K tile (32 keys x 64 d), row-major, async global->LDS
        {
            const bf16_t* src = Kb + (size_t)(kb + kkey) * rs + kseg;
            async_ld_b128(&Ks[kkey][kseg], src);
            if (kb + 32 < S) __builtin_prefetch(src + 32 * rs, 0, 1);
        }
        // stage V tile transposed: Vts[d][key]
        {
            Pack8 pk;
            const bf16_t* vp = Vb + (size_t)(kb + vks) * rs + vd;
            #pragma unroll
            for (int i = 0; i < 8; ++i)
                pk.h[i] = vp[(size_t)i * rs];
            *((u32x4*)&Vts[vd][vks]) = pk.u;
            if (kb + 32 < S) __builtin_prefetch(vp + 32 * rs, 0, 1);
        }
        wait_async0();
        __syncthreads();

        // --- scores: 16 queries x 32 keys, accumulate over d = 2 x 32 ---
        v8f s[2];
        #pragma unroll
        for (int j = 0; j < 2; ++j) {
            v8f a = {0.f,0.f,0.f,0.f,0.f,0.f,0.f,0.f};
            #pragma unroll
            for (int kk = 0; kk < 2; ++kk) {
                Frag32B fb;   // B-frag: lane -> key j*16+laneLo, d-slice {0..7,16..23} / {8..15,24..31}
                const bf16_t* p = &Ks[j * 16 + laneLo][kk * 32 + hi * 8];
                fb.u[0] = *((const u32x4*)p);
                fb.u[1] = *((const u32x4*)(p + 16));
                a = WMMA_BF16(qf[kk].v, fb.v, a);
            }
            s[j] = a;
        }
        #pragma unroll
        for (int j = 0; j < 2; ++j)
            #pragma unroll
            for (int r = 0; r < 8; ++r) s[j][r] *= scale;

        // --- online softmax: row reductions across 16-lane halves (wave32) ---
        float alpha[8];
        #pragma unroll
        for (int r = 0; r < 8; ++r) {
            float v = fmaxf(s[0][r], s[1][r]);
            v = fmaxf(v, __shfl_xor(v, 1, 32));
            v = fmaxf(v, __shfl_xor(v, 2, 32));
            v = fmaxf(v, __shfl_xor(v, 4, 32));
            v = fmaxf(v, __shfl_xor(v, 8, 32));
            float mnew = fmaxf(mrow[r], v);
            alpha[r] = exp2f((mrow[r] - mnew) * log2e);
            mrow[r]  = mnew;
        }
        #pragma unroll
        for (int j = 0; j < 2; ++j)
            #pragma unroll
            for (int r = 0; r < 8; ++r)
                s[j][r] = exp2f((s[j][r] - mrow[r]) * log2e);
        #pragma unroll
        for (int r = 0; r < 8; ++r) {
            float v = s[0][r] + s[1][r];
            v += __shfl_xor(v, 1, 32);
            v += __shfl_xor(v, 2, 32);
            v += __shfl_xor(v, 4, 32);
            v += __shfl_xor(v, 8, 32);
            lrow[r] = lrow[r] * alpha[r] + v;
        }
        #pragma unroll
        for (int j = 0; j < 4; ++j)
            #pragma unroll
            for (int r = 0; r < 8; ++r) o[j][r] *= alpha[r];

        // --- P: C-layout regs -> LDS -> A-layout frag ---
        #pragma unroll
        for (int j = 0; j < 2; ++j)
            #pragma unroll
            for (int r = 0; r < 8; ++r)
                Ps[wave][r + hi * 8][j * 16 + laneLo] = (bf16_t)s[j][r];
        __syncthreads();

        Frag32B pf;
        {
            const bf16_t* p = &Ps[wave][laneLo][hi * 8];
            pf.u[0] = *((const u32x4*)p);
            pf.u[1] = *((const u32x4*)(p + 16));
        }
        // --- O += P @ V (N = d, 4 x 16) ---
        #pragma unroll
        for (int jd = 0; jd < 4; ++jd) {
            Frag32B fv;
            const bf16_t* p = &Vts[jd * 16 + laneLo][hi * 8];
            fv.u[0] = *((const u32x4*)p);
            fv.u[1] = *((const u32x4*)(p + 16));
            o[jd] = WMMA_BF16(pf.v, fv.v, o[jd]);
        }
    }

    // --- epilogue: normalize by l, store bf16 ---
    #pragma unroll
    for (int jd = 0; jd < 4; ++jd) {
        int d = jd * 16 + laneLo;
        #pragma unroll
        for (int r = 0; r < 8; ++r) {
            int row = q0 + wave * 16 + r + hi * 8;
            float v = o[jd][r] / lrow[r];
            attnO[(size_t)(b * S + row) * 1024 + h * 64 + d] = (bf16_t)v;
        }
    }
}

// ---------- launcher ----------
extern "C" void kernel_launch(void* const* d_in, const int* in_sizes, int n_in,
                              void* d_out, int out_size, void* d_ws, size_t ws_size,
                              hipStream_t stream) {
    const float* x    = (const float*)d_in[0];
    const float* Wqkv = (const float*)d_in[1];
    const float* bqkv = (const float*)d_in[2];
    const float* Wo   = (const float*)d_in[3];
    const float* bo   = (const float*)d_in[4];
    float* out = (float*)d_out;

    const int Bb = 2, S = 2048, H = 1024;
    const int M  = Bb * S;          // 4096
    const int N1 = 3 * H;           // 3072

    char* ws = (char*)d_ws;
    bf16_t* xw    = (bf16_t*)ws;                      // 4096*1024
    bf16_t* wqkvw = (bf16_t*)(ws + 8388608);          // 1024*3072
    bf16_t* wow   = (bf16_t*)(ws + 8388608 + 6291456);// 1024*1024
    bf16_t* qkvw  = (bf16_t*)(ws + 8388608 + 6291456 + 2097152);            // 4096*3072
    bf16_t* attnw = (bf16_t*)(ws + 8388608 + 6291456 + 2097152 + 25165824); // 4096*1024

    // 1) fp32 -> bf16
    cvt_f32_to_bf16<<<(M * H) / 256, 256, 0, stream>>>(x, xw, M * H);
    cvt_f32_to_bf16<<<(H * N1) / 256, 256, 0, stream>>>(Wqkv, wqkvw, H * N1);
    cvt_f32_to_bf16<<<(H * H) / 256, 256, 0, stream>>>(Wo, wow, H * H);

    // 2) QKV projection: [4096,1024] x [1024,3072] + bqkv -> bf16
    gemm_wmma_bf16<false><<<dim3(N1 / 64, M / 128), 256, 0, stream>>>(
        xw, wqkvw, bqkv, nullptr, qkvw, M, N1, H);

    // 3) fused flash attention per (b, head)
    attn_flash_wmma<<<dim3(S / 128, Bb * 16), 256, 0, stream>>>(qkvw, attnw, S);

    // 4) output projection: [4096,1024] x [1024,1024] + bo -> fp32 out
    gemm_wmma_bf16<true><<<dim3(H / 64, M / 128), 256, 0, stream>>>(
        attnw, wow, bo, out, nullptr, M, H, H);
}
